// Seq2Seq_62096637166411
// MI455X (gfx1250) — matrix operand order repaired and checked by reading, not compiled, not run on hardware
//
#include <hip/hip_runtime.h>
#include <hip/hip_bf16.h>
#include <math.h>

typedef __bf16 bf16;
typedef __attribute__((ext_vector_type(16))) __bf16 v16bf;
typedef __attribute__((ext_vector_type(8)))  float  v8f;

#define VOCAB 32000
#define DMODEL 1024
#define DATT 256
#define CLOC 32
#define KLOC 31
#define BSZ 16
#define ESQ 512
#define DSQ 128

// ---------- WMMA fragment helpers (CDNA5 16x16x32 bf16, wave32) ----------
// A 16x32 (MxK): lane = half*16 + m ; element j -> k = half*16 + j
__device__ __forceinline__ v16bf load_a_frag(const bf16* __restrict__ A, int lda,
                                             int m0, int k0, int lane) {
  int m  = m0 + (lane & 15);
  int kh = (lane >> 4) * 16;
  const bf16* p = A + (size_t)m * lda + k0 + kh;
  v16bf r;
#pragma unroll
  for (int j = 0; j < 16; ++j) r[j] = p[j];
  return r;
}

// B 32x16 (KxN) built from row-major W (N,K):  B[k][n] = W[n0+n][k0+k]
// lane = half*16 + n ; element j -> k = half*16 + j   (16 contiguous bf16 per lane)
__device__ __forceinline__ v16bf load_b_fragT(const bf16* __restrict__ W, int ldw,
                                              int n0, int k0, int lane) {
  const bf16* p = W + (size_t)(n0 + (lane & 15)) * ldw + k0 + (lane >> 4) * 16;
  v16bf r;
#pragma unroll
  for (int j = 0; j < 16; ++j) r[j] = p[j];
  return r;
}

__device__ __forceinline__ v8f wmma_bf16(v16bf a, v16bf b, v8f c) {
  return __builtin_amdgcn_wmma_f32_16x16x32_bf16(false, a, false, b, (short)0, c,
                                                 false, false);
}

__device__ __forceinline__ float sigm(float x) { return 1.f / (1.f + __expf(-x)); }

// ---------- utility kernels ----------
__global__ void k_cvt_bf16(const float* __restrict__ s, bf16* __restrict__ d, long n) {
  long i = (long)blockIdx.x * blockDim.x + threadIdx.x;
  long stride = (long)gridDim.x * blockDim.x;
  for (; i < n; i += stride) d[i] = (bf16)s[i];
}

__global__ void k_zero(float* __restrict__ p, int n) {
  int i = blockIdx.x * blockDim.x + threadIdx.x;
  if (i < n) p[i] = 0.f;
}

// ---------- enc_proj: (16*512, 256) = enc(8192,1024) @ We^T(1024,256) ----------
__global__ void __launch_bounds__(256)
k_enc_proj(const bf16* __restrict__ encb, const bf16* __restrict__ Web,
           float* __restrict__ out) {
  int lane = threadIdx.x & 31, wave = threadIdx.x >> 5;
  int m0 = blockIdx.x * 16;
  for (int nt = wave; nt < DATT / 16; nt += 8) {
    v8f acc = {};
    for (int k0 = 0; k0 < DMODEL; k0 += 32) {
      v16bf a = load_a_frag(encb, DMODEL, m0, k0, lane);
      v16bf b = load_b_fragT(Web, DMODEL, nt * 16, k0, lane);
      acc = wmma_bf16(a, b, acc);
    }
    int n = nt * 16 + (lane & 15);
    int mb = m0 + 8 * (lane >> 4);
#pragma unroll
    for (int r = 0; r < 8; ++r) out[(size_t)(mb + r) * DATT + n] = acc[r];
  }
}

// ---------- x = emb[tok]*sqrt(D) + ctx ----------
__global__ void k_x(const float* __restrict__ emb, const int* __restrict__ tgt,
                    const float* __restrict__ ctx, float* __restrict__ x,
                    bf16* __restrict__ xbf, int t) {
  int i = blockIdx.x * blockDim.x + threadIdx.x;  // 16*1024
  int b = i >> 10, d = i & 1023;
  int tok = tgt[b * DSQ + t];
  float v = emb[(size_t)tok * DMODEL + d] * 32.0f + ctx[i];
  x[i] = v;
  xbf[i] = (bf16)v;
}

// ---------- fused LSTM layer: gates via 8 WMMA-accums / k-iter, then cell ----------
__global__ void __launch_bounds__(256)
k_lstm(const bf16* __restrict__ a1, const bf16* __restrict__ a2in,
       const bf16* __restrict__ Wih, const bf16* __restrict__ Whh,
       const float* __restrict__ bih, const float* __restrict__ bhh,
       float* __restrict__ c, float* __restrict__ h, bf16* __restrict__ hbf_out) {
  int lane = threadIdx.x & 31, wave = threadIdx.x >> 5;
  int tile = blockIdx.x * 8 + wave;  // 64 hidden-unit tiles
  int n0 = tile * 16;
  v8f acc[4] = {{}, {}, {}, {}};
  for (int k0 = 0; k0 < DMODEL; k0 += 32) {
    v16bf va = load_a_frag(a1, DMODEL, 0, k0, lane);
    v16bf vh = load_a_frag(a2in, DMODEL, 0, k0, lane);
#pragma unroll
    for (int g = 0; g < 4; ++g) {
      v16bf b1 = load_b_fragT(Wih, DMODEL, g * DMODEL + n0, k0, lane);
      acc[g] = wmma_bf16(va, b1, acc[g]);
      v16bf b2 = load_b_fragT(Whh, DMODEL, g * DMODEL + n0, k0, lane);
      acc[g] = wmma_bf16(vh, b2, acc[g]);
    }
  }
  int n = n0 + (lane & 15);
  int mb = 8 * (lane >> 4);
  float bi = bih[n] + bhh[n];
  float bf_ = bih[DMODEL + n] + bhh[DMODEL + n];
  float bg = bih[2 * DMODEL + n] + bhh[2 * DMODEL + n];
  float bo = bih[3 * DMODEL + n] + bhh[3 * DMODEL + n];
#pragma unroll
  for (int r = 0; r < 8; ++r) {
    size_t idx = (size_t)(mb + r) * DMODEL + n;
    float gi = acc[0][r] + bi, gf = acc[1][r] + bf_;
    float gg = acc[2][r] + bg, go = acc[3][r] + bo;
    float cn = sigm(gf) * c[idx] + sigm(gi) * tanhf(gg);
    float hn = sigm(go) * tanhf(cn);
    c[idx] = cn; h[idx] = hn; hbf_out[idx] = (bf16)hn;
  }
}

// ---------- q = h1 @ Wq^T : (16,256), K=1024 ----------
__global__ void __launch_bounds__(256)
k_q(const bf16* __restrict__ h1bf, const bf16* __restrict__ Wqb, float* __restrict__ q) {
  int lane = threadIdx.x & 31, wave = threadIdx.x >> 5;
  int nt = blockIdx.x * 8 + wave;  // 16 tiles
  if (nt >= DATT / 16) return;
  v8f acc = {};
  for (int k0 = 0; k0 < DMODEL; k0 += 32) {
    v16bf a = load_a_frag(h1bf, DMODEL, 0, k0, lane);
    v16bf b = load_b_fragT(Wqb, DMODEL, nt * 16, k0, lane);
    acc = wmma_bf16(a, b, acc);
  }
  int n = nt * 16 + (lane & 15);
  int mb = 8 * (lane >> 4);
#pragma unroll
  for (int r = 0; r < 8; ++r) q[(size_t)(mb + r) * DATT + n] = acc[r];
}

// ---------- location conv over previous attn; stores transposed bf16 (b,e,c) ----------
__global__ void k_conv(const float* __restrict__ attn, const float* __restrict__ cw,
                       const float* __restrict__ cb, bf16* __restrict__ fT) {
  int i = blockIdx.x * blockDim.x + threadIdx.x;  // 16*512*32
  int c = i & 31;
  int e = (i >> 5) & 511;
  int b = i >> 14;
  float s = cb[c];
#pragma unroll
  for (int k = 0; k < KLOC; ++k) {
    int p = e + k - 15;
    if (p >= 0 && p < ESQ) s += attn[b * ESQ + p] * cw[c * KLOC + k];
  }
  fT[i] = (bf16)s;  // i = (b*512 + e)*32 + c
}

// ---------- energy: e=tanh(q + enc_proj + f@Wf^T + b); score = e . wv ----------
__global__ void __launch_bounds__(256)
k_energy(const bf16* __restrict__ fT, const bf16* __restrict__ Wfb,
         const float* __restrict__ q, const float* __restrict__ encproj,
         const float* __restrict__ b_att, const float* __restrict__ wv,
         float* __restrict__ score) {
  int lane = threadIdx.x & 31, wave = threadIdx.x >> 5;
  int gw = blockIdx.x * 8 + wave;  // 16 b * 32 e-tiles = 512 waves
  int b = gw >> 5;
  int m0 = (gw & 31) * 16;
  v16bf a = load_a_frag(fT + (size_t)b * ESQ * CLOC, CLOC, m0, 0, lane);  // K=32 exact
  int mb = 8 * (lane >> 4);
  float part[8] = {0, 0, 0, 0, 0, 0, 0, 0};
  for (int at = 0; at < DATT / 16; ++at) {
    int a0 = at * 16;
    v16bf bb = load_b_fragT(Wfb, CLOC, a0, 0, lane);
    v8f acc = {};
    acc = wmma_bf16(a, bb, acc);
    int ai = a0 + (lane & 15);
    float wva = wv[ai], ba = b_att[ai], qa = q[b * DATT + ai];
#pragma unroll
    for (int r = 0; r < 8; ++r) {
      int e = m0 + mb + r;
      float v = tanhf(qa + encproj[((size_t)b * ESQ + e) * DATT + ai] + acc[r] + ba);
      part[r] += v * wva;
    }
  }
#pragma unroll
  for (int r = 0; r < 8; ++r) {  // reduce over the 16 lanes of each half-wave
    part[r] += __shfl_xor(part[r], 1, 32);
    part[r] += __shfl_xor(part[r], 2, 32);
    part[r] += __shfl_xor(part[r], 4, 32);
    part[r] += __shfl_xor(part[r], 8, 32);
  }
  if ((lane & 15) == 0) {
#pragma unroll
    for (int r = 0; r < 8; ++r) score[b * ESQ + m0 + mb + r] = part[r];
  }
}

// ---------- masked softmax, ctx = attn@enc, z_bf = bf16(ctx + h1) ----------
__global__ void __launch_bounds__(256)
k_softmax_ctx(const float* __restrict__ score, const int* __restrict__ src_mask,
              const float* __restrict__ enc, const float* __restrict__ h1,
              float* __restrict__ attn, float* __restrict__ ctx,
              bf16* __restrict__ zbf) {
  __shared__ float sa[ESQ];
  __shared__ float red[256];
  int b = blockIdx.x, tid = threadIdx.x;
  float lmax = -1e30f;
  for (int e = tid; e < ESQ; e += 256) {
    float v = (src_mask[b * ESQ + e] == 0) ? -1e9f : score[b * ESQ + e];
    sa[e] = v;
    lmax = fmaxf(lmax, v);
  }
  red[tid] = lmax; __syncthreads();
  for (int s = 128; s > 0; s >>= 1) {
    if (tid < s) red[tid] = fmaxf(red[tid], red[tid + s]);
    __syncthreads();
  }
  float mx = red[0]; __syncthreads();
  float lsum = 0.f;
  for (int e = tid; e < ESQ; e += 256) {
    float v = __expf(sa[e] - mx); sa[e] = v; lsum += v;
  }
  red[tid] = lsum; __syncthreads();
  for (int s = 128; s > 0; s >>= 1) {
    if (tid < s) red[tid] += red[tid + s];
    __syncthreads();
  }
  float inv = 1.f / red[0]; __syncthreads();
  for (int e = tid; e < ESQ; e += 256) { sa[e] *= inv; attn[b * ESQ + e] = sa[e]; }
  __syncthreads();
  for (int d = tid; d < DMODEL; d += 256) {
    float s = 0.f;
    const float* ep = enc + (size_t)b * ESQ * DMODEL + d;
    for (int e = 0; e < ESQ; ++e) s += sa[e] * ep[(size_t)e * DMODEL];
    size_t idx = (size_t)b * DMODEL + d;
    ctx[idx] = s;
    zbf[idx] = (bf16)(s + h1[idx]);
  }
}

// ---------- logits: z(16,1024) @ out_W^T(1024,32000) -> d_out[b, t, v] ----------
__global__ void __launch_bounds__(256)
k_logits(const bf16* __restrict__ zbf, const bf16* __restrict__ outWb,
         float* __restrict__ out, int t) {
  int lane = threadIdx.x & 31, wave = threadIdx.x >> 5;
  int nt = blockIdx.x * 8 + wave;  // 2000 N-tiles
  int n0 = nt * 16;
  v8f acc = {};
  for (int k0 = 0; k0 < DMODEL; k0 += 32) {
    v16bf a = load_a_frag(zbf, DMODEL, 0, k0, lane);
    v16bf b = load_b_fragT(outWb, DMODEL, n0, k0, lane);
    acc = wmma_bf16(a, b, acc);
  }
  int n = n0 + (lane & 15);
  int mb = 8 * (lane >> 4);
#pragma unroll
  for (int r = 0; r < 8; ++r)
    out[((size_t)(mb + r) * DSQ + t) * VOCAB + n] = acc[r];
}

// ============================ host ============================
extern "C" void kernel_launch(void* const* d_in, const int* in_sizes, int n_in,
                              void* d_out, int out_size, void* d_ws, size_t ws_size,
                              hipStream_t stream) {
  (void)in_sizes; (void)n_in; (void)out_size; (void)ws_size;
  const float* src_seq = (const float*)d_in[0];
  const int*   src_mask = (const int*)d_in[1];
  const int*   tgt_seq = (const int*)d_in[2];
  const float* emb_W = (const float*)d_in[3];
  const float* Wih0 = (const float*)d_in[4];
  const float* Whh0 = (const float*)d_in[5];
  const float* bih0 = (const float*)d_in[6];
  const float* bhh0 = (const float*)d_in[7];
  const float* Wih1 = (const float*)d_in[8];
  const float* Whh1 = (const float*)d_in[9];
  const float* bih1 = (const float*)d_in[10];
  const float* bhh1 = (const float*)d_in[11];
  const float* conv_w = (const float*)d_in[12];
  const float* conv_b = (const float*)d_in[13];
  const float* Wq = (const float*)d_in[14];
  const float* We = (const float*)d_in[15];
  const float* Wf = (const float*)d_in[16];
  const float* b_att = (const float*)d_in[17];
  const float* wv = (const float*)d_in[18];
  const float* out_W = (const float*)d_in[19];
  float* out = (float*)d_out;

  char* ws = (char*)d_ws;
  size_t off = 0;
  auto alloc = [&](size_t bytes) -> void* {
    void* p = ws + off;
    off += (bytes + 255) & ~(size_t)255;
    return p;
  };
  // bf16 weight mirrors (one-time per launch; makes working set L2-resident)
  bf16* Wih0b = (bf16*)alloc((size_t)4 * DMODEL * DMODEL * 2);
  bf16* Whh0b = (bf16*)alloc((size_t)4 * DMODEL * DMODEL * 2);
  bf16* Wih1b = (bf16*)alloc((size_t)4 * DMODEL * DMODEL * 2);
  bf16* Whh1b = (bf16*)alloc((size_t)4 * DMODEL * DMODEL * 2);
  bf16* outWb = (bf16*)alloc((size_t)VOCAB * DMODEL * 2);
  bf16* encb  = (bf16*)alloc((size_t)BSZ * ESQ * DMODEL * 2);
  bf16* Wqb   = (bf16*)alloc((size_t)DATT * DMODEL * 2);
  bf16* Web   = (bf16*)alloc((size_t)DATT * DMODEL * 2);
  bf16* Wfb   = (bf16*)alloc((size_t)DATT * CLOC * 2);
  float* encproj = (float*)alloc((size_t)BSZ * ESQ * DATT * 4);
  // zero-initialized state block (contiguous; sizes all multiples of 256B)
  float* h0  = (float*)alloc(BSZ * DMODEL * 4);
  float* c0  = (float*)alloc(BSZ * DMODEL * 4);
  float* h1  = (float*)alloc(BSZ * DMODEL * 4);
  float* c1  = (float*)alloc(BSZ * DMODEL * 4);
  float* ctx = (float*)alloc(BSZ * DMODEL * 4);
  float* x   = (float*)alloc(BSZ * DMODEL * 4);
  float* attn  = (float*)alloc(BSZ * ESQ * 4);
  float* score = (float*)alloc(BSZ * ESQ * 4);
  bf16* h0bfA = (bf16*)alloc(BSZ * DMODEL * 2);
  bf16* h0bfB = (bf16*)alloc(BSZ * DMODEL * 2);
  bf16* h1bfA = (bf16*)alloc(BSZ * DMODEL * 2);
  bf16* h1bfB = (bf16*)alloc(BSZ * DMODEL * 2);
  bf16* xbf   = (bf16*)alloc(BSZ * DMODEL * 2);
  bf16* zbf   = (bf16*)alloc(BSZ * DMODEL * 2);
  // not needing init
  float* qbuf = (float*)alloc(BSZ * DATT * 4);
  bf16* fTbf  = (bf16*)alloc((size_t)BSZ * ESQ * CLOC * 2);

  auto cvt = [&](const float* s, bf16* d, long n) {
    int blocks = (int)((n + 1023) / 1024);
    if (blocks > 4096) blocks = 4096;
    if (blocks < 1) blocks = 1;
    hipLaunchKernelGGL(k_cvt_bf16, dim3(blocks), dim3(256), 0, stream, s, d, n);
  };
  cvt(Wih0, Wih0b, (long)4 * DMODEL * DMODEL);
  cvt(Whh0, Whh0b, (long)4 * DMODEL * DMODEL);
  cvt(Wih1, Wih1b, (long)4 * DMODEL * DMODEL);
  cvt(Whh1, Whh1b, (long)4 * DMODEL * DMODEL);
  cvt(out_W, outWb, (long)VOCAB * DMODEL);
  cvt(src_seq, encb, (long)BSZ * ESQ * DMODEL);
  cvt(Wq, Wqb, (long)DATT * DMODEL);
  cvt(We, Web, (long)DATT * DMODEL);
  cvt(Wf, Wfb, (long)DATT * CLOC);

  // zero h0..zbf (contiguous): 6*64KB f32 + 2*32KB f32 + 6*32KB bf16
  int zero_floats = (6 * BSZ * DMODEL * 4 + 2 * BSZ * ESQ * 4 + 6 * BSZ * DMODEL * 2) / 4;
  hipLaunchKernelGGL(k_zero, dim3((zero_floats + 255) / 256), dim3(256), 0, stream,
                     h0, zero_floats);

  // enc_proj (one-time): 512 row-tiles
  hipLaunchKernelGGL(k_enc_proj, dim3(BSZ * ESQ / 16), dim3(256), 0, stream,
                     encb, Web, encproj);

  for (int t = 0; t < DSQ; ++t) {
    bf16* h0_in  = (t & 1) ? h0bfB : h0bfA;
    bf16* h0_out = (t & 1) ? h0bfA : h0bfB;
    bf16* h1_in  = (t & 1) ? h1bfB : h1bfA;
    bf16* h1_out = (t & 1) ? h1bfA : h1bfB;

    hipLaunchKernelGGL(k_x, dim3(BSZ * DMODEL / 256), dim3(256), 0, stream,
                       emb_W, tgt_seq, ctx, x, xbf, t);
    hipLaunchKernelGGL(k_lstm, dim3(8), dim3(256), 0, stream,
                       xbf, h0_in, Wih0b, Whh0b, bih0, bhh0, c0, h0, h0_out);
    hipLaunchKernelGGL(k_lstm, dim3(8), dim3(256), 0, stream,
                       h0_out, h1_in, Wih1b, Whh1b, bih1, bhh1, c1, h1, h1_out);
    hipLaunchKernelGGL(k_q, dim3(2), dim3(256), 0, stream, h1_out, Wqb, qbuf);
    hipLaunchKernelGGL(k_conv, dim3(BSZ * ESQ * CLOC / 256), dim3(256), 0, stream,
                       attn, conv_w, conv_b, fTbf);
    hipLaunchKernelGGL(k_energy, dim3(64), dim3(256), 0, stream,
                       fTbf, Wfb, qbuf, encproj, b_att, wv, score);
    hipLaunchKernelGGL(k_softmax_ctx, dim3(BSZ), dim3(256), 0, stream,
                       score, src_mask, src_seq, h1, attn, ctx, zbf);
    hipLaunchKernelGGL(k_logits, dim3(VOCAB / 16 / 8), dim3(256), 0, stream,
                       zbf, outWb, out, t);
  }

  // outputs 2 & 3 of the tuple: src_seq and src_mask, appended flat
  size_t o1 = (size_t)BSZ * DSQ * VOCAB;
  size_t o2 = (size_t)BSZ * ESQ * DMODEL;
  hipMemcpyAsync(out + o1, src_seq, o2 * sizeof(float), hipMemcpyDeviceToDevice, stream);
  hipMemcpyAsync(out + o1 + o2, src_mask, (size_t)BSZ * ESQ * sizeof(int),
                 hipMemcpyDeviceToDevice, stream);
}
